// MultiHeadAttention_56392920597047
// MI455X (gfx1250) — compile-verified, matrix operand order checked by reading, not run on hardware
//
#include <hip/hip_runtime.h>

// ---------------------------------------------------------------------------
// MI455X (gfx1250) fused multi-head attention.
//
// Roofline: ~52 GFLOP dense vs ~80 MB traffic (fits in 192MB L2) -> compute
// bound. fp32->f16 once at the projections, every matmul stage on
// v_wmma_f32_16x16x32_f16 (f32 accumulate), f16 intermediates, flash
// softmax (S x S scores never hit HBM), and TDM (tensor_load_to_lds)
// double-buffered K/V tile staging in the attention kernel.
// ---------------------------------------------------------------------------

#define DEV_INLINE __device__ __forceinline__

typedef _Float16 half8  __attribute__((ext_vector_type(8)));
typedef _Float16 half16 __attribute__((ext_vector_type(16)));
typedef float    float8 __attribute__((ext_vector_type(8)));
typedef unsigned int uint32x4 __attribute__((ext_vector_type(4)));
typedef int          int32x8  __attribute__((ext_vector_type(8)));
typedef int          int32x4  __attribute__((ext_vector_type(4)));

#if defined(__has_builtin)
#if __has_builtin(__builtin_amdgcn_tensor_load_to_lds) && \
    __has_builtin(__builtin_amdgcn_s_wait_tensorcnt)
#define HAVE_TDM 1
#endif
#endif

constexpr int Bn  = 4;
constexpr int Sn  = 1024;
constexpr int Dn  = 1024;
constexpr int Hn  = 16;
constexpr int DHn = 64;
constexpr int Mtot = Bn * Sn;          // 4096 token rows

// ---------------------------------------------------------------------------
// WMMA operand fragment loader.
// 16-bit A-matrix 16x32 layout (CDNA5 ISA 7.12.2):
//   lanes 0-15  (halfSel=0): dwords 0-3 = K 0..7,  dwords 4-7 = K 16..23
//   lanes 16-31 (halfSel=1): dwords 0-3 = K 8..15, dwords 4-7 = K 24..31
// rowPtr = 32-element K-slice for this lane's row (A) / column (B^T).
// Both 8-half groups are 16B aligned -> two b128 loads (global or ds).
// ---------------------------------------------------------------------------
DEV_INLINE half16 load_frag(const _Float16* rowPtr, int halfSel) {
  const _Float16* p = rowPtr + halfSel * 8;
  half8 lo = *(const half8*)(p);
  half8 hi = *(const half8*)(p + 16);
  half16 r;
#pragma unroll
  for (int i = 0; i < 8; ++i) { r[i] = lo[i]; r[i + 8] = hi[i]; }
  return r;
}

DEV_INLINE float8 wmma16(half16 a, half16 b, float8 c) {
  return __builtin_amdgcn_wmma_f32_16x16x32_f16(
      false, a, false, b, (short)0, c, false, false);
}

#ifdef HAVE_TDM
// ---------------------------------------------------------------------------
// TDM 2D tile load: global (row-major, stride_elems f16 per row) -> LDS.
// D# per CDNA5 ISA ch.8: group0 = {count/flags, lds_addr, global_addr, type},
// group1 = {data_size + LDS padding, tensor dims == tile dims (no OOB),
// tile dims, dim0 stride}. pad_enable inserts (pad_amount+1) dwords every
// (1<<pad_interval)*8 bytes so padded LDS rows stay 16B aligned and
// bank-spread for the b128 fragment reads. This toolchain exposes the 6-arg
// builtin: (g0 u32x4, g1 i32x8, g2 i32x4, g3 i32x4, i32x8, i32 cpol).
// ---------------------------------------------------------------------------
DEV_INLINE void tdm_load_2d_f16(unsigned lds_off, const _Float16* gptr,
                                unsigned tile_d0, unsigned tile_d1,
                                unsigned stride_elems, unsigned pad_interval,
                                unsigned pad_amount) {
  unsigned long long ga = (unsigned long long)(uintptr_t)gptr;
  uint32x4 g0;
  g0[0] = 1u;                                             // count=1, user D#
  g0[1] = lds_off;                                        // LDS byte address
  g0[2] = (unsigned)ga;                                   // global_addr lo
  g0[3] = (unsigned)((ga >> 32) & 0x01FFFFFFu) | (2u << 30);  // hi25 | type=2
  int32x8 g1;
  g1[0] = (int)((1u << 16) |                              // data_size = 2B
                (1u << 20) |                              // pad_enable
                (pad_interval << 22) | (pad_amount << 25));
  g1[1] = (int)((tile_d0 & 0xFFFFu) << 16);               // tensor_dim0 lo16
  g1[2] = (int)(((tile_d0 >> 16) & 0xFFFFu) |             // tensor_dim0 hi16
                ((tile_d1 & 0xFFFFu) << 16));             // tensor_dim1 lo16
  g1[3] = (int)(((tile_d1 >> 16) & 0xFFFFu) |             // tensor_dim1 hi16
                ((tile_d0 & 0xFFFFu) << 16));             // tile_dim0
  g1[4] = (int)(tile_d1 & 0xFFFFu);                       // tile_dim1, d2=0
  g1[5] = (int)stride_elems;                              // dim0 stride lo32
  g1[6] = 0;
  g1[7] = 0;
  int32x4 gz4 = {0, 0, 0, 0};
  int32x8 gz8 = {0, 0, 0, 0, 0, 0, 0, 0};
  __builtin_amdgcn_tensor_load_to_lds(g0, g1, gz4, gz4, gz8, 0);
}
#endif

// ---------------------------------------------------------------------------
// GEMM: Y = relu(X[4096x1024] @ W[1024x1024] + bias), WMMA-tiled.
// Block 128(M) x 64(N), 256 thr = 8 waves, wave = 16x64 strip (A fragment
// reused across 4 B tiles). K staged in 32-chunks through LDS as f16.
// OMODE: 0 = f16 row-major, 1 = f16 head-transposed [b][h][dh][s] (for V),
//        2 = f32 row-major (final output).
// ---------------------------------------------------------------------------
template <bool XF16, int OMODE>
__global__ __launch_bounds__(256) void gemm_bias_relu(
    const void* __restrict__ Xv, const float* __restrict__ W,
    const float* __restrict__ bias, void* __restrict__ Yv) {
  __shared__ __align__(16) _Float16 Xs[128][40];   // 80B row stride
  __shared__ __align__(16) _Float16 WsT[64][40];   // W chunk, transposed

  const int tid     = threadIdx.x;
  const int wave    = tid >> 5;
  const int lane    = tid & 31;
  const int m       = lane & 15;
  const int halfSel = lane >> 4;
  const int blockN  = blockIdx.x * 64;
  const int blockM  = blockIdx.y * 128;

  const _Float16* Xh = (const _Float16*)Xv;
  const float*    Xf = (const float*)Xv;

  float8 acc[4];
#pragma unroll
  for (int t = 0; t < 4; ++t)
#pragma unroll
    for (int i = 0; i < 8; ++i) acc[t][i] = 0.0f;

  for (int kc = 0; kc < Dn / 32; ++kc) {
    // ---- stage X block: 128 rows x 32 k (coalesced) ----
#pragma unroll
    for (int i = 0; i < 16; ++i) {
      int e = tid + 256 * i;
      int r = e >> 5, c = e & 31;
      size_t gi = (size_t)(blockM + r) * Dn + kc * 32 + c;
      if (XF16) Xs[r][c] = Xh[gi];
      else      Xs[r][c] = (_Float16)Xf[gi];
    }
    // ---- stage W chunk transposed: 32 k x 64 n ----
#pragma unroll
    for (int i = 0; i < 8; ++i) {
      int e = tid + 256 * i;
      int kr = e >> 6, nc = e & 63;
      WsT[nc][kr] = (_Float16)W[(size_t)(kc * 32 + kr) * Dn + blockN + nc];
    }
    if (kc + 1 < Dn / 32) {   // prefetch next chunk into caches
      __builtin_prefetch(&W[(size_t)(kc * 32 + 32 + (tid >> 3)) * Dn + blockN],
                         0, 1);
      if (XF16)
        __builtin_prefetch(
            &Xh[(size_t)(blockM + (tid >> 1)) * Dn + kc * 32 + 32], 0, 1);
      else
        __builtin_prefetch(
            &Xf[(size_t)(blockM + (tid >> 1)) * Dn + kc * 32 + 32], 0, 1);
    }
    __syncthreads();

    half16 a = load_frag(&Xs[wave * 16 + m][0], halfSel);
#pragma unroll
    for (int t = 0; t < 4; ++t) {
      half16 b = load_frag(&WsT[t * 16 + m][0], halfSel);
      acc[t] = wmma16(a, b, acc[t]);
    }
    __syncthreads();
  }

  // ---- epilogue: bias + relu + store ----
#pragma unroll
  for (int t = 0; t < 4; ++t) {
#pragma unroll
    for (int r = 0; r < 8; ++r) {
      int row = blockM + wave * 16 + halfSel * 8 + r;
      int col = blockN + t * 16 + m;
      float v = acc[t][r] + bias[col];
      v = fmaxf(v, 0.0f);
      if (OMODE == 0) {
        ((_Float16*)Yv)[(size_t)row * Dn + col] = (_Float16)v;
      } else if (OMODE == 1) {
        int b_ = row >> 10, s_ = row & 1023;
        int h_ = col >> 6,  dh = col & 63;
        ((_Float16*)Yv)[(((size_t)(b_ * Hn + h_) * DHn + dh) << 10) + s_] =
            (_Float16)v;
      } else {
        ((float*)Yv)[(size_t)row * Dn + col] = v;
      }
    }
  }
}

// ---------------------------------------------------------------------------
// Flash attention: one wave per 16-row Q tile of one (b,h); streams keys in
// 32-key blocks. K/V tiles staged to LDS by the Tensor Data Mover, double
// buffered, synchronized with s_wait_tensorcnt (up to 2 tiles in flight
// while the next pair is issued). Scores: 4 WMMA; online softmax with
// shfl_xor reductions in the 16-lane C-layout groups; P re-shaped through a
// small LDS tile into A-operand layout; O += P*V: 4 WMMA.
// ---------------------------------------------------------------------------
__global__ __launch_bounds__(32) void flash_attn(
    const _Float16* __restrict__ qp, const _Float16* __restrict__ kp,
    const _Float16* __restrict__ vpT, _Float16* __restrict__ ctx) {
#ifdef HAVE_TDM
  // padded LDS tiles written by TDM: K row = 64 halves + 8 pad = 144B,
  // V row = 32 halves + 8 pad = 80B (both 16B aligned).
  __shared__ __align__(16) _Float16 Klds[2][32][72];
  __shared__ __align__(16) _Float16 Vlds[2][64][40];
#endif
  __shared__ __align__(16) _Float16 Plds[16][40];

  const int qBlock  = blockIdx.x & 63;
  const int bh      = blockIdx.x >> 6;
  const int h       = bh & 15;
  const int b       = bh >> 4;
  const int lane    = threadIdx.x;
  const int m       = lane & 15;
  const int halfSel = lane >> 4;
  const int rowBase = b * Sn + qBlock * 16;

  half16 qa[2];
#pragma unroll
  for (int c = 0; c < 2; ++c)
    qa[c] = load_frag(qp + (size_t)(rowBase + m) * Dn + h * DHn + c * 32,
                      halfSel);

  float8 o[4];
  float mrow[8], lrow[8];
#pragma unroll
  for (int t = 0; t < 4; ++t)
#pragma unroll
    for (int i = 0; i < 8; ++i) o[t][i] = 0.0f;
#pragma unroll
  for (int r = 0; r < 8; ++r) { mrow[r] = -1.0e30f; lrow[r] = 0.0f; }

  const float scale = 0.125f;               // 1/sqrt(DH)
  constexpr int nJ = Sn / 32;

#ifdef HAVE_TDM
  auto issue_tiles = [&](int j, int buf) {
    int kb = j * 32;
    // K tile: 32 token rows x 64 features, row stride Dn
    tdm_load_2d_f16((unsigned)(uintptr_t)&Klds[buf][0][0],
                    kp + (size_t)(b * Sn + kb) * Dn + h * DHn,
                    /*d0=*/64, /*d1=*/32, /*stride=*/Dn,
                    /*pad_interval=*/4, /*pad_amount=*/3);
    // V tile (head-transposed): 64 dh rows x 32 keys, row stride Sn
    tdm_load_2d_f16((unsigned)(uintptr_t)&Vlds[buf][0][0],
                    vpT + ((size_t)bh * DHn) * Sn + kb,
                    /*d0=*/32, /*d1=*/64, /*stride=*/Sn,
                    /*pad_interval=*/3, /*pad_amount=*/3);
  };
  issue_tiles(0, 0);
#endif

  for (int j = 0; j < nJ; ++j) {
    const int keyBase = j * 32;
#ifdef HAVE_TDM
    const int buf = j & 1;
    if (j + 1 < nJ) {
      issue_tiles(j + 1, buf ^ 1);
      __builtin_amdgcn_s_wait_tensorcnt(2);   // tile j's two loads done
    } else {
      __builtin_amdgcn_s_wait_tensorcnt(0);
    }
#endif

    // ---- scores: two 16-key tiles, K-dim 64 = 2 WMMA each ----
    float8 s0, s1;
#pragma unroll
    for (int i = 0; i < 8; ++i) { s0[i] = 0.0f; s1[i] = 0.0f; }
#pragma unroll
    for (int c = 0; c < 2; ++c) {
#ifdef HAVE_TDM
      half16 kb0 = load_frag(&Klds[buf][m][c * 32], halfSel);
      half16 kb1 = load_frag(&Klds[buf][16 + m][c * 32], halfSel);
#else
      half16 kb0 = load_frag(
          kp + (size_t)(b * Sn + keyBase + m) * Dn + h * DHn + c * 32, halfSel);
      half16 kb1 = load_frag(
          kp + (size_t)(b * Sn + keyBase + 16 + m) * Dn + h * DHn + c * 32,
          halfSel);
#endif
      s0 = wmma16(qa[c], kb0, s0);
      s1 = wmma16(qa[c], kb1, s1);
    }

    // ---- online softmax update per C-layout row ----
#pragma unroll
    for (int r = 0; r < 8; ++r) {
      float a0 = s0[r] * scale;
      float a1 = s1[r] * scale;
      float mt = fmaxf(a0, a1);
#pragma unroll
      for (int off = 1; off < 16; off <<= 1)
        mt = fmaxf(mt, __shfl_xor(mt, off, 32));
      float mnew = fmaxf(mrow[r], mt);
      float corr = __expf(mrow[r] - mnew);
      mrow[r] = mnew;
      lrow[r] *= corr;
#pragma unroll
      for (int t = 0; t < 4; ++t) o[t][r] *= corr;
      float p0 = __expf(a0 - mnew);
      float p1 = __expf(a1 - mnew);
      float ps = p0 + p1;
#pragma unroll
      for (int off = 1; off < 16; off <<= 1)
        ps += __shfl_xor(ps, off, 32);
      lrow[r] += ps;
      Plds[halfSel * 8 + r][m]      = (_Float16)p0;
      Plds[halfSel * 8 + r][16 + m] = (_Float16)p1;
    }
    __syncthreads();

    // ---- O += P * V ----
    half16 pa = load_frag(&Plds[m][0], halfSel);
#pragma unroll
    for (int t = 0; t < 4; ++t) {
      int dh = t * 16 + m;
#ifdef HAVE_TDM
      half16 vb = load_frag(&Vlds[buf][dh][0], halfSel);
#else
      half16 vb = load_frag(
          vpT + (((size_t)(bh * DHn + dh)) << 10) + keyBase, halfSel);
#endif
      o[t] = wmma16(pa, vb, o[t]);
    }
    __syncthreads();
  }

  // ---- epilogue: normalize, write context (f16 row-major) ----
#pragma unroll
  for (int t = 0; t < 4; ++t) {
#pragma unroll
    for (int r = 0; r < 8; ++r) {
      int row = rowBase + halfSel * 8 + r;
      int col = h * DHn + t * 16 + m;
      ctx[(size_t)row * Dn + col] = (_Float16)(o[t][r] / lrow[r]);
    }
  }
}

// ---------------------------------------------------------------------------
extern "C" void kernel_launch(void* const* d_in, const int* in_sizes, int n_in,
                              void* d_out, int out_size, void* d_ws,
                              size_t ws_size, hipStream_t stream) {
  const float* q  = (const float*)d_in[0];
  const float* k  = (const float*)d_in[1];
  const float* v  = (const float*)d_in[2];
  const float* Wq = (const float*)d_in[3];
  const float* bq = (const float*)d_in[4];
  const float* Wk = (const float*)d_in[5];
  const float* bk = (const float*)d_in[6];
  const float* Wv = (const float*)d_in[7];
  const float* bv = (const float*)d_in[8];
  const float* Wo = (const float*)d_in[9];
  const float* bo = (const float*)d_in[10];

  // workspace: 4 x (4096x1024) f16 buffers = 32 MB
  _Float16* qp  = (_Float16*)d_ws;
  _Float16* kp  = qp  + (size_t)Mtot * Dn;
  _Float16* vpT = kp  + (size_t)Mtot * Dn;   // head-transposed [b][h][dh][s]
  _Float16* ctx = vpT + (size_t)Mtot * Dn;

  dim3 ggrid(Dn / 64, Mtot / 128);
  dim3 gblk(256);

  gemm_bias_relu<false, 0><<<ggrid, gblk, 0, stream>>>(q, Wq, bq, qp);
  gemm_bias_relu<false, 0><<<ggrid, gblk, 0, stream>>>(k, Wk, bk, kp);
  gemm_bias_relu<false, 1><<<ggrid, gblk, 0, stream>>>(v, Wv, bv, vpT);

  flash_attn<<<dim3(Bn * Hn * (Sn / 16)), dim3(32), 0, stream>>>(qp, kp, vpT,
                                                                 ctx);

  gemm_bias_relu<true, 2><<<ggrid, gblk, 0, stream>>>(ctx, Wo, bo,
                                                      (float*)d_out);
}
